// SelectiveSSM1D_58136677319324
// MI455X (gfx1250) — matrix operand order repaired and checked by reading, not compiled, not run on hardware
//
#include <hip/hip_runtime.h>
#include <hip/hip_bf16.h>
#include <math.h>

// ---------------------------------------------------------------------------
// SelectiveSSM1D for MI455X (gfx1250, wave32, WMMA).
// B=8, L=4096, D=768, N=16, R=8.  M = B*L = 32768.
//
// y = hs @ W_out^T + b_out + x @ W_skip^T + b_skip
// hs from linear recurrence h <- h*diag(a) + h@(V U^T) + sigmoid(xWg+bg)*(xWi+bi)
//
// Skip GEMM (38.7 of 41 GFLOP) runs on the bf16 16x16x32 WMMA pipe with
// hi/lo split precision (3 bf16 WMMAs per K=32, fp32 accumulate).
// x tiles are staged to LDS with GLOBAL_LOAD_ASYNC_TO_LDS_B128 (ASYNCcnt),
// pipelined one K-block ahead of the WMMA stream.
// ---------------------------------------------------------------------------

#define BB   8
#define LL   4096
#define DD   768
#define NN   16
#define RR   8
#define MM   (BB * LL)          // 32768 rows
#define KKW  (DD / 2)           // 384 packed K-pair dwords per W column

typedef float  v2f   __attribute__((ext_vector_type(2)));
typedef float  v8f   __attribute__((ext_vector_type(8)));
typedef int    v4i   __attribute__((ext_vector_type(4)));
typedef __bf16 v16bf __attribute__((ext_vector_type(16)));

static __device__ __forceinline__ v8f wmma_f32_k4(v2f a, v2f b, v8f c) {
    return __builtin_amdgcn_wmma_f32_16x16x4_f32(
        false, a, false, b, (short)0, c, false, false);
}
static __device__ __forceinline__ v8f wmma_bf16_k32(v16bf a, v16bf b, v8f c) {
    return __builtin_amdgcn_wmma_f32_16x16x32_bf16(
        false, a, false, b, (short)0, c, false, false);
}

union FragBF { v4i i[2]; v16bf v; };

static __device__ __forceinline__ unsigned short f2bf(float f) {
    unsigned u = __float_as_uint(f);
    return (unsigned short)((u + 0x7fffu + ((u >> 16) & 1u)) >> 16);
}
static __device__ __forceinline__ float bf2f(unsigned short h) {
    return __uint_as_float(((unsigned)h) << 16);
}
static __device__ __forceinline__ unsigned pack2(unsigned short lo, unsigned short hi) {
    return (unsigned)lo | ((unsigned)hi << 16);
}
static __device__ __forceinline__ float sigm(float v) {
    return __builtin_amdgcn_rcpf(1.0f + __expf(-v));
}

// global -> LDS async copy of 16B, tracked by ASYNCcnt (GV addressing mode)
static __device__ __forceinline__ void async_g2l_b128(const void* gptr, void* lds) {
    unsigned           ldsoff = (unsigned)(unsigned long long)lds;   // addr[31:0] = LDS offset
    unsigned long long gaddr  = (unsigned long long)gptr;
    asm volatile("global_load_async_to_lds_b128 %0, %1, off"
                 :: "v"(ldsoff), "v"(gaddr) : "memory");
}
static __device__ __forceinline__ void wait_async0() {
    asm volatile("s_wait_asynccnt 0x0" ::: "memory");
}

// Workspace layout (float slots)
#define OFS_WPROJT   0                           // [768][32] f32: cols 0-15 W_in, 16-31 W_gate
#define OFS_WOUTT    (OFS_WPROJT + DD * 32)      // [16][768] f32
#define OFS_WSKIP_HI (OFS_WOUTT + NN * DD)       // [768][384] u32: bf16-hi K-pairs, col-major
#define OFS_WSKIP_LO (OFS_WSKIP_HI + DD * KKW)   // [768][384] u32: bf16-lo K-pairs
#define OFS_GU       (OFS_WSKIP_LO + DD * KKW)   // [M][16] f32
#define OFS_HS       (OFS_GU + MM * NN)          // [M][16] f32

// ---------------------------------------------------------------------------
// Kernel 0: weight repacking (transposes + bf16 hi/lo B-fragment pack).
// ---------------------------------------------------------------------------
__global__ void ssm_prep(const float* __restrict__ W_in,
                         const float* __restrict__ W_gate,
                         const float* __restrict__ W_out,
                         const float* __restrict__ W_skip,
                         float* __restrict__ ws) {
    int idx = blockIdx.x * blockDim.x + threadIdx.x;

    if (idx < DD * KKW) {            // B[k][d] = W_skip[d][k], packed K-pairs
        int d = idx / KKW, kk = idx - d * KKW;
        float w0 = W_skip[d * DD + 2 * kk];
        float w1 = W_skip[d * DD + 2 * kk + 1];
        unsigned short h0 = f2bf(w0), h1 = f2bf(w1);
        unsigned short l0 = f2bf(w0 - bf2f(h0)), l1 = f2bf(w1 - bf2f(h1));
        unsigned* whi = (unsigned*)(ws + OFS_WSKIP_HI);
        unsigned* wlo = (unsigned*)(ws + OFS_WSKIP_LO);
        whi[d * KKW + kk] = pack2(h0, h1);
        wlo[d * KKW + kk] = pack2(l0, l1);
    }
    if (idx < DD * 32) {             // WprojT[k][n]
        int k = idx >> 5, n = idx & 31;
        float v = (n < NN) ? W_in[n * DD + k] : W_gate[(n - NN) * DD + k];
        ws[OFS_WPROJT + k * 32 + n] = v;
    }
    if (idx < NN * DD) {             // WoutT[n][d]
        int n = idx / DD, d = idx - n * DD;
        ws[OFS_WOUTT + n * DD + d] = W_out[d * NN + n];
    }
}

// ---------------------------------------------------------------------------
// Kernel 1: gu = sigmoid(x@Wg^T + bg) * (x@Wi^T + bi), fp32 K=4 WMMA chain.
// ---------------------------------------------------------------------------
__global__ void ssm_proj(const float* __restrict__ x,
                         const float* __restrict__ ws,
                         const float* __restrict__ b_in,
                         const float* __restrict__ b_gate,
                         float* __restrict__ gu) {
    const float* WprojT = ws + OFS_WPROJT;
    int wave = blockIdx.x * (blockDim.x >> 5) + (threadIdx.x >> 5);
    int lane = threadIdx.x & 31;
    int m    = lane & 15;
    int half = lane >> 4;
    int rowbase = wave * 16;

    const float2* xr = (const float2*)(x + (size_t)(rowbase + m) * DD);

    v8f accU = {}; v8f accG = {};
#pragma unroll 4
    for (int k0 = 0; k0 < DD; k0 += 4) {
        float2 av = xr[(k0 >> 1) + half];
        v2f A; A.x = av.x; A.y = av.y;
        int kr = (k0 + 2 * half) * 32;
        v2f Bu, Bg;
        Bu.x = WprojT[kr + m];      Bu.y = WprojT[kr + 32 + m];
        Bg.x = WprojT[kr + 16 + m]; Bg.y = WprojT[kr + 48 + m];
        accU = wmma_f32_k4(A, Bu, accU);
        accG = wmma_f32_k4(A, Bg, accG);
    }

    float bi = b_in[m];
    float bg = b_gate[m];
#pragma unroll
    for (int i = 0; i < 8; i++) {
        float u = accU[i] + bi;
        float g = sigm(accG[i] + bg);
        int row = rowbase + i + 8 * half;
        gu[(size_t)row * NN + m] = g * u;
    }
}

// ---------------------------------------------------------------------------
// Kernel 2: sequential scan.  One block, 4 waves = 8 batch states; the 16x16
// transition matvec uses 16 intra-wave shuffles (no LDS, no barriers).
// ---------------------------------------------------------------------------
__global__ void ssm_scan(const float* __restrict__ gu,
                         const float* __restrict__ a_logit,
                         const float* __restrict__ U,
                         const float* __restrict__ V,
                         float* __restrict__ hs) {
    int w    = threadIdx.x >> 5;
    int lane = threadIdx.x & 31;
    int b    = 2 * w + (lane >> 4);
    int n    = lane & 15;

    float LTcol[NN];
#pragma unroll
    for (int mm = 0; mm < NN; mm++) {
        float s = 0.f;
#pragma unroll
        for (int r = 0; r < RR; r++) s += V[mm * RR + r] * U[n * RR + r];
        LTcol[mm] = s;
    }
    float a_n = sigm(a_logit[n]);

    const float* gub = gu + (size_t)b * LL * NN;
    float*       hsb = hs + (size_t)b * LL * NN;

    float h = 0.0f;
    for (int t = 0; t < LL; t++) {
        float g = gub[t * NN + n];
        float s = 0.0f;
#pragma unroll
        for (int mm = 0; mm < NN; mm++)
            s = fmaf(__shfl(h, mm, 16), LTcol[mm], s);
        h = fmaf(h, a_n, g + s);
        hsb[t * NN + n] = h;
    }
}

// ---------------------------------------------------------------------------
// Kernel 3: y = hs @ WoutT + x @ WskipT + (b_out + b_skip).
// 256 thr (8 waves) per block, 32 rows x 256 cols (halves W L2 traffic vs
// 16-row blocks).  x K-blocks are fetched with async global->LDS b128 copies
// one block ahead (overlapping the WMMA stream), then split to bf16 hi/lo.
// grid = (M/32, D/256).
// ---------------------------------------------------------------------------
#define KB   32
#define ROWS 32
__global__ void ssm_out(const float* __restrict__ x,
                        const float* __restrict__ ws,
                        const float* __restrict__ b_out,
                        const float* __restrict__ b_skip,
                        float* __restrict__ y) {
    const float*    WoutT = ws + OFS_WOUTT;
    const float*    hs    = ws + OFS_HS;
    const unsigned* whi   = (const unsigned*)(ws + OFS_WSKIP_HI);
    const unsigned* wlo   = (const unsigned*)(ws + OFS_WSKIP_LO);

    __shared__ __align__(16) float          sX[ROWS * KB];     // fp32 async landing tile
    __shared__ __align__(16) unsigned short sAhi[ROWS * KB];
    __shared__ __align__(16) unsigned short sAlo[ROWS * KB];
    __shared__ __align__(16) float          sH[ROWS * NN];

    int rowbase = blockIdx.x * ROWS;
    int wv      = threadIdx.x >> 5;
    int lane    = threadIdx.x & 31;
    int m       = lane & 15;
    int half    = lane >> 4;
    int rowtile = wv >> 2;                       // 0..1
    int colbase = blockIdx.y * 256 + (wv & 3) * 64;
    int arow    = rowtile * 16 + m;              // A-row within block tile

    // per-thread staging slot: 256 thr x 16B = full 32x32 fp32 tile
    int r  = threadIdx.x >> 3;                   // 0..31
    int c4 = (threadIdx.x & 7) * 4;
    float*       sxp   = &sX[r * KB + c4];
    const float* gsrc0 = &x[(size_t)(rowbase + r) * DD + c4];

    v8f acc[4] = {{}, {}, {}, {}};

    // preload K-block 0
    async_g2l_b128(gsrc0, sxp);

    for (int k0 = 0; k0 < DD; k0 += KB) {
        wait_async0();
        __syncthreads();                         // tile visible to all waves
        {   // split fp32 tile -> bf16 hi/lo tiles
            float4 xv = *(const float4*)sxp;
            unsigned short h0 = f2bf(xv.x), h1 = f2bf(xv.y),
                           h2 = f2bf(xv.z), h3 = f2bf(xv.w);
            unsigned short l0 = f2bf(xv.x - bf2f(h0)), l1 = f2bf(xv.y - bf2f(h1)),
                           l2 = f2bf(xv.z - bf2f(h2)), l3 = f2bf(xv.w - bf2f(h3));
            unsigned* dh = (unsigned*)sAhi + ((r * KB + c4) >> 1);
            unsigned* dl = (unsigned*)sAlo + ((r * KB + c4) >> 1);
            dh[0] = pack2(h0, h1); dh[1] = pack2(h2, h3);
            dl[0] = pack2(l0, l1); dl[1] = pack2(l2, l3);
        }
        __syncthreads();                         // sX drained, bf16 tiles ready

        // overlap next K-block's global->LDS copy with the WMMAs below
        if (k0 + KB < DD) async_g2l_b128(gsrc0 + k0 + KB, sxp);

        // A fragments (16-bit layout: K = 8*half..+7 then 16+8*half..+7)
        FragBF Ahi, Alo;
        Ahi.i[0] = *(const v4i*)&sAhi[arow * KB + 8 * half];
        Ahi.i[1] = *(const v4i*)&sAhi[arow * KB + 16 + 8 * half];
        Alo.i[0] = *(const v4i*)&sAlo[arow * KB + 8 * half];
        Alo.i[1] = *(const v4i*)&sAlo[arow * KB + 16 + 8 * half];

        int kk0 = (k0 >> 1) + 4 * half;          // packed dword K base for this half
#pragma unroll
        for (int j = 0; j < 4; j++) {
            int col = colbase + 16 * j + m;
            const v4i* ph = (const v4i*)(whi + (size_t)col * KKW + kk0);
            const v4i* pl = (const v4i*)(wlo + (size_t)col * KKW + kk0);
            FragBF Bhi, Blo;
            Bhi.i[0] = ph[0]; Bhi.i[1] = ph[2];
            Blo.i[0] = pl[0]; Blo.i[1] = pl[2];
            acc[j] = wmma_bf16_k32(Ahi.v, Bhi.v, acc[j]);
            acc[j] = wmma_bf16_k32(Ahi.v, Blo.v, acc[j]);
            acc[j] = wmma_bf16_k32(Alo.v, Bhi.v, acc[j]);
        }
    }

    // ---- out-proj: K = 16 from hs, fp32 WMMA ----
    __syncthreads();
    if (threadIdx.x < 128) {
        int r2  = threadIdx.x >> 2;              // 0..31
        int c42 = (threadIdx.x & 3) * 4;
        *(float4*)&sH[r2 * NN + c42] =
            *(const float4*)&hs[(size_t)(rowbase + r2) * NN + c42];
    }
    __syncthreads();
#pragma unroll
    for (int kk = 0; kk < NN; kk += 4) {
        float2 av = *(const float2*)&sH[arow * NN + kk + 2 * half];
        v2f A; A.x = av.x; A.y = av.y;
        const float* brow = WoutT + (size_t)(kk + 2 * half) * DD;
#pragma unroll
        for (int j = 0; j < 4; j++) {
            int col = colbase + 16 * j + m;
            v2f Bv; Bv.x = brow[col]; Bv.y = brow[DD + col];
            acc[j] = wmma_f32_k4(A, Bv, acc[j]);
        }
    }

    // ---- bias + store ----
#pragma unroll
    for (int j = 0; j < 4; j++) {
        int col = colbase + 16 * j + m;
        float bias = b_out[col] + b_skip[col];
#pragma unroll
        for (int i = 0; i < 8; i++) {
            int row = rowbase + rowtile * 16 + i + 8 * half;
            y[(size_t)row * DD + col] = acc[j][i] + bias;
        }
    }
}

// ---------------------------------------------------------------------------
extern "C" void kernel_launch(void* const* d_in, const int* in_sizes, int n_in,
                              void* d_out, int out_size, void* d_ws, size_t ws_size,
                              hipStream_t stream) {
    const float* x       = (const float*)d_in[0];
    const float* a_logit = (const float*)d_in[1];
    const float* U       = (const float*)d_in[2];
    const float* V       = (const float*)d_in[3];
    const float* W_in    = (const float*)d_in[4];
    const float* b_in    = (const float*)d_in[5];
    const float* W_gate  = (const float*)d_in[6];
    const float* b_gate  = (const float*)d_in[7];
    const float* W_out   = (const float*)d_in[8];
    const float* b_out   = (const float*)d_in[9];
    const float* W_skip  = (const float*)d_in[10];
    const float* b_skip  = (const float*)d_in[11];

    float* ws = (float*)d_ws;
    float* y  = (float*)d_out;
    float* gu = ws + OFS_GU;
    float* hs = ws + OFS_HS;

    ssm_prep<<<(DD * KKW + 255) / 256, 256, 0, stream>>>(W_in, W_gate, W_out, W_skip, ws);
    ssm_proj<<<MM / 16 / 4, 128, 0, stream>>>(x, ws, b_in, b_gate, gu);
    ssm_scan<<<1, 128, 0, stream>>>(gu, a_logit, U, V, hs);
    ssm_out<<<dim3(MM / ROWS, DD / 256), 256, 0, stream>>>(x, ws, b_out, b_skip, y);
}
